// SupplyPlanningTGNN_45148696215819
// MI455X (gfx1250) — matrix operand order repaired and checked by reading, not compiled, not run on hardware
//
#include <hip/hip_runtime.h>
#include <hip/hip_bf16.h>

// ---------------- problem constants ----------------
#define BB 4
#define TT 12
#define NN 10000
#define EE 160000
#define TD 10
#define HID 64
#define HEADS 2
#define HID2 128          // HEADS*HID
#define G3 192            // 3*HID
#define ETOT (EE + NN)    // edges + self loops

typedef __attribute__((ext_vector_type(16))) _Float16 v16h;
typedef __attribute__((ext_vector_type(8)))  float    v8f;

// ---------------- small device helpers ----------------
__device__ __forceinline__ float sigmoidf_(float x) { return 1.f / (1.f + __expf(-x)); }
__device__ __forceinline__ float eluf_(float x)     { return x > 0.f ? x : (__expf(x) - 1.f); }
__device__ __forceinline__ float lrelu_(float x)    { return x > 0.f ? x : 0.2f * x; }

// float atomic max via int/uint ordering trick (works for all finite floats, init -inf)
__device__ __forceinline__ void atomicMaxF(float* addr, float val) {
    if (val >= 0.f) atomicMax((int*)addr, __float_as_int(val));
    else            atomicMin((unsigned int*)addr, (unsigned int)__float_as_int(val));
}

__device__ __forceinline__ void edge_sd(const int* ei, int e, int& s, int& d) {
    if (e < EE) { s = ei[e]; d = ei[EE + e]; }
    else        { s = e - EE; d = e - EE; }     // appended self loops
}

// ---- CDNA5 async memory <-> LDS (ASYNCcnt-tracked, no VGPR data path) ----
__device__ __forceinline__ void async_load_b128(unsigned lds_off, const float* gaddr) {
    asm volatile("global_load_async_to_lds_b128 %0, %1, off"
                 :: "v"(lds_off), "v"((unsigned long long)(uintptr_t)gaddr)
                 : "memory");
}
__device__ __forceinline__ void async_store_b128(unsigned lds_off, float* gaddr) {
    asm volatile("global_store_async_from_lds_b128 %1, %0, off"
                 :: "v"(lds_off), "v"((unsigned long long)(uintptr_t)gaddr)
                 : "memory");
}
__device__ __forceinline__ void wait_async0() {
    asm volatile("s_wait_asynccnt 0x0" ::: "memory");
}

// ============================================================
// 1. sop_h[n,h] = sop_emb@sop_W + sop_b + lat@lat_W + lat_b
// ============================================================
__global__ void k_soph(const float* __restrict__ sop, const float* __restrict__ sopW,
                       const float* __restrict__ sopB, const float* __restrict__ lat,
                       const float* __restrict__ latW, const float* __restrict__ latB,
                       float* __restrict__ sop_h) {
    int idx = blockIdx.x * blockDim.x + threadIdx.x;
    if (idx >= NN * HID) return;
    int n = idx >> 6, h = idx & 63;
    float v = sopB[h] + latB[h];
    const float* sr = sop + (size_t)n * 64;
    for (int k = 0; k < 64; ++k) v += sr[k] * sopW[k * HID + h];
    const float* lr = lat + (size_t)n * 6;
    for (int k = 0; k < 6; ++k)  v += lr[k] * latW[k * HID + h];
    sop_h[idx] = v;
}

// ============================================================
// 2. node_h[b,t,n,h] = x_temporal@tx_W + tx_b + sop_h[n,h]
// ============================================================
__global__ void k_nodeh(const float* __restrict__ xt, const float* __restrict__ txW,
                        const float* __restrict__ txB, const float* __restrict__ sop_h,
                        float* __restrict__ node_h) {
    size_t idx = (size_t)blockIdx.x * blockDim.x + threadIdx.x;
    if (idx >= (size_t)BB * TT * NN * HID) return;
    int h = idx & 63;
    size_t rest = idx >> 6;           // bt*N + n
    int n = (int)(rest % NN);
    float v = txB[h] + sop_h[n * HID + h];
    const float* xr = xt + rest * TD;
    for (int k = 0; k < TD; ++k) v += xr[k] * txW[k * HID + h];
    node_h[idx] = v;
}

// ============================================================
// 3. Generic batched WMMA GEMM: C[z] = A[z] @ W   (f16 in, f32 acc)
//    one wave -> one 16x16 output tile
// ============================================================
__global__ __launch_bounds__(32)
void k_gemm16(const float* __restrict__ A0, long sA, int lda,
              const float* __restrict__ W, int ldw,
              float* __restrict__ C0, long sC, int ldc, int K) {
    int lane = threadIdx.x;
    int m0 = blockIdx.x * 16, n0 = blockIdx.y * 16;
    const float* A = A0 + (size_t)blockIdx.z * sA;
    float*       C = C0 + (size_t)blockIdx.z * sC;
    int rA = lane & 15, hiA = lane >> 4, hiB = lane >> 4;
    v8f acc = {};
    for (int kc = 0; kc < K; kc += 32) {
        __builtin_prefetch(&A[(size_t)(m0 + rA) * lda + kc + 32], 0, 1);
        v16h a, bf;
#pragma unroll
        for (int j = 0; j < 16; ++j) {
            int k = kc + hiA * 8 + (j & 7) + ((j >> 3) << 4);
            a[j] = (_Float16)A[(size_t)(m0 + rA) * lda + k];
        }
#pragma unroll
        for (int j = 0; j < 16; ++j) {
            int k = kc + hiB * 8 + (j & 7) + ((j >> 3) << 4);
            bf[j] = (_Float16)W[(size_t)k * ldw + n0 + (lane & 15)];
        }
        acc = __builtin_amdgcn_wmma_f32_16x16x32_f16(false, a, false, bf,
                                                     (short)0, acc, false, false);
    }
#pragma unroll
    for (int i = 0; i < 8; ++i) {
        int row = (lane < 16) ? i : i + 8;
        C[(size_t)(m0 + row) * ldc + n0 + (lane & 15)] = acc[i];
    }
}

// ============================================================
// 4. GAT edge-phase kernels
// ============================================================
__global__ void k_gatinit(float* __restrict__ gout, float* __restrict__ m,
                          float* __restrict__ den) {
    size_t idx = (size_t)blockIdx.x * blockDim.x + threadIdx.x;
    if (idx >= (size_t)BB * NN * HID2) return;
    gout[idx] = 0.f;
    if (idx < (size_t)BB * NN * HEADS) { m[idx] = -3.0e38f; den[idx] = 0.f; }
}

__global__ void k_logits(const int* __restrict__ ei, const float* __restrict__ xl,
                         const float* __restrict__ xr, const float* __restrict__ att,
                         float* __restrict__ pex, float* __restrict__ m) {
    size_t idx = (size_t)blockIdx.x * blockDim.x + threadIdx.x;
    if (idx >= (size_t)BB * ETOT * HEADS) return;
    int h = idx & 1; size_t r = idx >> 1;
    int e = (int)(r % ETOT), b = (int)(r / ETOT);
    int s, d; edge_sd(ei, e, s, d);
    const float* xls = xl + ((size_t)b * NN + s) * HID2 + h * HID;
    const float* xrd = xr + ((size_t)b * NN + d) * HID2 + h * HID;
    const float* at  = att + h * HID;
    float lg = 0.f;
    for (int k = 0; k < HID; ++k) lg += at[k] * lrelu_(xls[k] + xrd[k]);
    pex[idx] = lg;
    atomicMaxF(&m[((size_t)b * NN + d) * HEADS + h], lg);
}

__global__ void k_pexp(const int* __restrict__ ei, const float* __restrict__ m,
                       float* __restrict__ pex, float* __restrict__ den) {
    size_t idx = (size_t)blockIdx.x * blockDim.x + threadIdx.x;
    if (idx >= (size_t)BB * ETOT * HEADS) return;
    int h = idx & 1; size_t r = idx >> 1;
    int e = (int)(r % ETOT), b = (int)(r / ETOT);
    int s, d; edge_sd(ei, e, s, d); (void)s;
    float p = __expf(pex[idx] - m[((size_t)b * NN + d) * HEADS + h]);
    pex[idx] = p;
    atomicAdd(&den[((size_t)b * NN + d) * HEADS + h], p);
}

__global__ void k_scatter(const int* __restrict__ ei, const float* __restrict__ xl,
                          const float* __restrict__ pex, const float* __restrict__ den,
                          float* __restrict__ gout) {
    size_t idx = (size_t)blockIdx.x * blockDim.x + threadIdx.x;
    if (idx >= (size_t)BB * ETOT * HID2) return;
    int c = idx & 127; size_t r = idx >> 7;
    int e = (int)(r % ETOT), b = (int)(r / ETOT);
    int h = c >> 6;
    int s, d; edge_sd(ei, e, s, d);
    float alpha = pex[((size_t)b * ETOT + e) * HEADS + h] /
                  (den[((size_t)b * NN + d) * HEADS + h] + 1e-16f);
    atomicAdd(&gout[((size_t)b * NN + d) * HID2 + c],
              xl[((size_t)b * NN + s) * HID2 + c] * alpha);
}

__global__ void k_post1(const float* __restrict__ gout, const float* __restrict__ bias,
                        float* __restrict__ h1) {
    size_t idx = (size_t)blockIdx.x * blockDim.x + threadIdx.x;
    if (idx >= (size_t)BB * NN * HID2) return;
    h1[idx] = eluf_(gout[idx] + bias[idx & 127]);
}

__global__ void k_post2(const float* __restrict__ gout, const float* __restrict__ bias,
                        float* __restrict__ sp) {
    size_t idx = (size_t)blockIdx.x * blockDim.x + threadIdx.x;
    if (idx >= (size_t)BB * NN * HID) return;
    size_t bn = idx >> 6; int d = idx & 63;
    float v = 0.5f * (gout[bn * HID2 + d] + gout[bn * HID2 + HID + d]) + bias[d];
    sp[idx] = eluf_(v);
}

// ============================================================
// 5. Pack GRU weights (Wih^T / Whh^T) into per-lane WMMA B-fragments (f16)
//    layout: (((mat*12+nt)*2+kc2)*32+lane)*16 + j ;  K = kc2*32+(lane>>4)*8+(j&7)+(j>>3)*16
// ============================================================
__global__ void k_packgru(const float* __restrict__ Wih, const float* __restrict__ Whh,
                          _Float16* __restrict__ wp) {
    int idx = blockIdx.x * blockDim.x + threadIdx.x;
    if (idx >= 2 * 12 * 2 * 32 * 16) return;
    int j = idx & 15, lane = (idx >> 4) & 31, kc2 = (idx >> 9) & 1;
    int v = idx >> 10, nt = v % 12, mat = v / 12;
    int n = nt * 16 + (lane & 15);            // output col in [0,192)
    int k = kc2 * 32 + (lane >> 4) * 8 + (j & 7) + ((j >> 3) << 4);  // K in [0,64)
    const float* W = mat ? Whh : Wih;         // (192,64) row-major; B[k][n] = W[n][k]
    wp[idx] = (_Float16)W[(size_t)n * HID + k];
}

// ============================================================
// 6. Fused GRU over T steps: one wave owns 16 rows, h lives in LDS.
//    x_t tiles are staged global->LDS via async-tensor path (ASYNCcnt),
//    h_T is written back LDS->global the same way.
// ============================================================
__global__ __launch_bounds__(32)
void k_gru(const float* __restrict__ node_h, const float* __restrict__ sp,
           const _Float16* __restrict__ wp, const float* __restrict__ bih,
           const float* __restrict__ bhh, float* __restrict__ th) {
    __shared__ __align__(16) float xbuf[16 * 64];
    __shared__ __align__(16) float hbuf[16 * 64];
    __shared__ __align__(16) float spb [16 * 64];
    __shared__ float gsum[16 * 128];
    __shared__ float gin [16 * 64];
    __shared__ float ghn [16 * 64];
    int lane = threadIdx.x;
    int r0 = blockIdx.x * 16;                 // global row tile (B*N rows, N%16==0)
    int b = r0 / NN, n0 = r0 % NN;
    int rA = lane & 15, hiA = lane >> 4;
    unsigned xoff = (unsigned)(uintptr_t)(void*)xbuf;   // LDS byte offset (low 32 bits)
    unsigned hoff = (unsigned)(uintptr_t)(void*)hbuf;

    for (int i = lane; i < 16 * 64; i += 32) {
        spb[i]  = sp[(size_t)(r0 + (i >> 6)) * HID + (i & 63)];
        hbuf[i] = 0.f;
    }
    __syncthreads();

    for (int t = 0; t < TT; ++t) {
        // --- async stage x_t tile (contiguous 4KB) into LDS, b128 per lane ---
        const float* xb = node_h + ((size_t)(b * TT + t) * NN + n0) * HID;
#pragma unroll
        for (int q = 0; q < 8; ++q) {
            int f = (q * 32 + lane) * 4;               // float index, 16B aligned
            async_load_b128(xoff + (unsigned)f * 4u, xb + f);
        }
        wait_async0();
        __syncthreads();

        for (int nt = 0; nt < 12; ++nt) {
            v8f gi = {}, gh = {};
#pragma unroll
            for (int kc2 = 0; kc2 < 2; ++kc2) {
                int kc = kc2 * 32;
                v16h ax, ah;
#pragma unroll
                for (int j = 0; j < 16; ++j) {
                    int k = kc + hiA * 8 + (j & 7) + ((j >> 3) << 4);
                    ax[j] = (_Float16)(xbuf[rA * 64 + k] + spb[rA * 64 + k]);
                    ah[j] = (_Float16)hbuf[rA * 64 + k];
                }
                const v16h bu = *(const v16h*)(wp + (((0 * 12 + nt) * 2 + kc2) * 32 + lane) * 16);
                const v16h bv = *(const v16h*)(wp + (((1 * 12 + nt) * 2 + kc2) * 32 + lane) * 16);
                gi = __builtin_amdgcn_wmma_f32_16x16x32_f16(false, ax, false, bu,
                                                            (short)0, gi, false, false);
                gh = __builtin_amdgcn_wmma_f32_16x16x32_f16(false, ah, false, bv,
                                                            (short)0, gh, false, false);
            }
            int col = nt * 16 + rA;
#pragma unroll
            for (int i = 0; i < 8; ++i) {
                int row = (lane < 16) ? i : i + 8;
                float vi = gi[i] + bih[col];
                float vh = gh[i] + bhh[col];
                if (nt < 8) gsum[row * 128 + col] = vi + vh;          // r,z parts
                else { gin[row * 64 + col - 128] = vi;                // n parts
                       ghn[row * 64 + col - 128] = vh; }
            }
        }
        __syncthreads();

        for (int q = 0; q < 32; ++q) {
            int flat = q * 32 + lane;
            int row = flat >> 6, col = flat & 63;
            float r = sigmoidf_(gsum[row * 128 + col]);
            float z = sigmoidf_(gsum[row * 128 + col + 64]);
            float nc = tanhf(gin[row * 64 + col] + r * ghn[row * 64 + col]);
            hbuf[flat] = (1.f - z) * nc + z * hbuf[flat];
        }
        __syncthreads();
    }

    // --- async write h_T tile LDS -> global ---
    float* tbase = th + (size_t)r0 * HID;
#pragma unroll
    for (int q = 0; q < 8; ++q) {
        int f = (q * 32 + lane) * 4;
        async_store_b128(hoff + (unsigned)f * 4u, tbase + f);
    }
    wait_async0();
}

// ============================================================
// 7. Five MLP heads -> d_out (B,N,5)
// ============================================================
__device__ float head_eval(const float* x, const float* W1, const float* b1,
                           const float* W2, const float* b2, int Hd, bool sig) {
    float o = b2[0];
    for (int u = 0; u < Hd; ++u) {
        float a = b1[u];
        for (int d = 0; d < HID; ++d) a += x[d] * W1[d * Hd + u];
        a = a > 0.f ? a : 0.f;
        o += a * W2[u];
    }
    return sig ? sigmoidf_(o) : (o > 0.f ? o : 0.f);
}

__global__ void k_heads(const float* __restrict__ th,
                        const float* eW1, const float* eb1, const float* eW2, const float* eb2,
                        const float* oW1, const float* ob1, const float* oW2, const float* ob2,
                        const float* aW1, const float* ab1, const float* aW2, const float* ab2,
                        const float* lW1, const float* lb1, const float* lW2, const float* lb2,
                        const float* cW1, const float* cb1, const float* cW2, const float* cb2,
                        float* __restrict__ out) {
    int idx = blockIdx.x * blockDim.x + threadIdx.x;
    if (idx >= BB * NN) return;
    float x[HID];
    const float* tr = th + (size_t)idx * HID;
    for (int d = 0; d < HID; ++d) x[d] = tr[d];
    float* o = out + (size_t)idx * 5;
    o[0] = head_eval(x, eW1, eb1, eW2, eb2, 16, true);
    o[1] = head_eval(x, oW1, ob1, oW2, ob2, 32, false);
    o[2] = head_eval(x, aW1, ab1, aW2, ab2, 16, true);
    o[3] = head_eval(x, lW1, lb1, lW2, lb2, 16, true);
    o[4] = head_eval(x, cW1, cb1, cW2, cb2, 16, true);
}

// ============================================================
extern "C" void kernel_launch(void* const* d_in, const int* in_sizes, int n_in,
                              void* d_out, int out_size, void* d_ws, size_t ws_size,
                              hipStream_t stream) {
    const float* x_temporal = (const float*)d_in[0];
    const float* sop_emb    = (const float*)d_in[1];
    const float* lat        = (const float*)d_in[2];
    const float* tx_W  = (const float*)d_in[3];  const float* tx_b  = (const float*)d_in[4];
    const float* sop_W = (const float*)d_in[5];  const float* sop_b = (const float*)d_in[6];
    const float* lat_W = (const float*)d_in[7];  const float* lat_b = (const float*)d_in[8];
    const float* c1_Wl = (const float*)d_in[9];  const float* c1_Wr = (const float*)d_in[10];
    const float* c1_att= (const float*)d_in[11]; const float* c1_b  = (const float*)d_in[12];
    const float* c2_Wl = (const float*)d_in[13]; const float* c2_Wr = (const float*)d_in[14];
    const float* c2_att= (const float*)d_in[15]; const float* c2_b  = (const float*)d_in[16];
    const float* gWih  = (const float*)d_in[17]; const float* gWhh  = (const float*)d_in[18];
    const float* gbih  = (const float*)d_in[19]; const float* gbhh  = (const float*)d_in[20];
    const int*   ei    = (const int*)d_in[41];

    float* ws = (float*)d_ws;
    float* node_h = ws;                               // 30,720,000
    float* sop_h  = node_h + (size_t)BB*TT*NN*HID;    //    640,000
    float* sp     = sop_h  + (size_t)NN*HID;          //  2,560,000
    float* th     = sp     + (size_t)BB*NN*HID;       //  2,560,000
    float* xl     = th     + (size_t)BB*NN*HID;       //  5,120,000
    float* xr     = xl     + (size_t)BB*NN*HID2;
    float* gout   = xr     + (size_t)BB*NN*HID2;
    float* h1     = gout   + (size_t)BB*NN*HID2;
    float* mbuf   = h1     + (size_t)BB*NN*HID2;      //     80,000
    float* dbuf   = mbuf   + (size_t)BB*NN*HEADS;
    float* pex    = dbuf   + (size_t)BB*NN*HEADS;     //  1,360,000
    _Float16* wpack = (_Float16*)(pex + (size_t)BB*ETOT*HEADS);

    float* out = (float*)d_out;

    // 1-2: node features
    k_soph<<<(NN*HID + 255)/256, 256, 0, stream>>>(sop_emb, sop_W, sop_b, lat, lat_W, lat_b, sop_h);
    {
        size_t tot = (size_t)BB*TT*NN*HID;
        k_nodeh<<<(unsigned)((tot + 255)/256), 256, 0, stream>>>(x_temporal, tx_W, tx_b, sop_h, node_h);
    }
    k_packgru<<<(2*12*2*32*16 + 255)/256, 256, 0, stream>>>(gWih, gWhh, wpack);

    dim3 ggrid(NN/16, HID2/16, BB);     // (625, 8, 4)
    size_t nEH = (size_t)BB*ETOT*HEADS, nEC = (size_t)BB*ETOT*HID2, nNC = (size_t)BB*NN*HID2;

    // ---- GAT layer 1 (A = node_h[:, T-1]) ----
    k_gemm16<<<ggrid, 32, 0, stream>>>(node_h + (size_t)(TT-1)*NN*HID, (long)TT*NN*HID, HID,
                                       c1_Wl, HID2, xl, (long)NN*HID2, HID2, HID);
    k_gemm16<<<ggrid, 32, 0, stream>>>(node_h + (size_t)(TT-1)*NN*HID, (long)TT*NN*HID, HID,
                                       c1_Wr, HID2, xr, (long)NN*HID2, HID2, HID);
    k_gatinit<<<(unsigned)((nNC + 255)/256), 256, 0, stream>>>(gout, mbuf, dbuf);
    k_logits <<<(unsigned)((nEH + 255)/256), 256, 0, stream>>>(ei, xl, xr, c1_att, pex, mbuf);
    k_pexp   <<<(unsigned)((nEH + 255)/256), 256, 0, stream>>>(ei, mbuf, pex, dbuf);
    k_scatter<<<(unsigned)((nEC + 255)/256), 256, 0, stream>>>(ei, xl, pex, dbuf, gout);
    k_post1  <<<(unsigned)((nNC + 255)/256), 256, 0, stream>>>(gout, c1_b, h1);

    // ---- GAT layer 2 (A = h1, K = 128) ----
    k_gemm16<<<ggrid, 32, 0, stream>>>(h1, (long)NN*HID2, HID2, c2_Wl, HID2,
                                       xl, (long)NN*HID2, HID2, HID2);
    k_gemm16<<<ggrid, 32, 0, stream>>>(h1, (long)NN*HID2, HID2, c2_Wr, HID2,
                                       xr, (long)NN*HID2, HID2, HID2);
    k_gatinit<<<(unsigned)((nNC + 255)/256), 256, 0, stream>>>(gout, mbuf, dbuf);
    k_logits <<<(unsigned)((nEH + 255)/256), 256, 0, stream>>>(ei, xl, xr, c2_att, pex, mbuf);
    k_pexp   <<<(unsigned)((nEH + 255)/256), 256, 0, stream>>>(ei, mbuf, pex, dbuf);
    k_scatter<<<(unsigned)((nEC + 255)/256), 256, 0, stream>>>(ei, xl, pex, dbuf, gout);
    k_post2  <<<(unsigned)(((size_t)BB*NN*HID + 255)/256), 256, 0, stream>>>(gout, c2_b, sp);

    // ---- fused GRU (sp added at fragment-pack time) ----
    k_gru<<<BB*NN/16, 32, 0, stream>>>(node_h, sp, wpack, gbih, gbhh, th);

    // ---- output heads ----
    k_heads<<<(BB*NN + 63)/64, 64, 0, stream>>>(th,
        (const float*)d_in[21], (const float*)d_in[22], (const float*)d_in[23], (const float*)d_in[24],
        (const float*)d_in[25], (const float*)d_in[26], (const float*)d_in[27], (const float*)d_in[28],
        (const float*)d_in[29], (const float*)d_in[30], (const float*)d_in[31], (const float*)d_in[32],
        (const float*)d_in[33], (const float*)d_in[34], (const float*)d_in[35], (const float*)d_in[36],
        (const float*)d_in[37], (const float*)d_in[38], (const float*)d_in[39], (const float*)d_in[40],
        out);
}
